// TransformerBlock_1434519077372
// MI455X (gfx1250) — matrix operand order repaired and checked by reading, not compiled, not run on hardware
//
#include <hip/hip_runtime.h>
#include <cstddef>

// ---------------------------------------------------------------------------
// Problem constants (reference: D_MODEL=1024, N_HEADS=16, D_FF=4096, B=2, T=2048)
// ---------------------------------------------------------------------------
#define TB      2
#define TT      2048
#define DM      1024
#define NH      16
#define HD      64
#define DFF     4096
#define MROWS   (TB * TT)          // 4096

typedef unsigned short u16;
typedef __attribute__((ext_vector_type(8)))  u16    u16x8;
typedef __attribute__((ext_vector_type(16))) __bf16 v16bf;
typedef __attribute__((ext_vector_type(8)))  float  v8f;

union Frag {            // one WMMA 16-half operand (8 VGPRs)
    v16bf b;
    u16x8 h[2];
};

__device__ inline u16 f2bf(float f) {          // round-to-nearest-even f32->bf16
    union { float f; unsigned u; } c; c.f = f;
    unsigned u = c.u;
    unsigned r = u + 0x7FFFu + ((u >> 16) & 1u);
    return (u16)(r >> 16);
}

__device__ inline float gelu_f(float x) {      // exact gelu (erf form)
    return 0.5f * x * (1.0f + erff(x * 0.70710678118f));
}

__device__ inline v8f wmma_bf16(const Frag& a, const Frag& b, v8f c) {
    // (neg_a, A, neg_b, B, c_mod, C, reuse_a, reuse_b)
    return __builtin_amdgcn_wmma_f32_16x16x32_bf16(false, a.b, false, b.b,
                                                   (short)0, c, false, false);
}

// gfx1250 async global->LDS copy (16B per lane), tracked by ASYNCcnt.
__device__ inline void async_g2l_b128(u16* lds_dst, const u16* gsrc) {
    unsigned l = (unsigned)(size_t)lds_dst;    // low 32 bits = LDS byte address
    asm volatile("global_load_async_to_lds_b128 %0, %1, off"
                 :: "v"(l), "v"(gsrc) : "memory");
}
__device__ inline void wait_async0() {
    asm volatile("s_wait_asynccnt 0x0" ::: "memory");
}

// ---------------------------------------------------------------------------
// Elementwise f32 -> bf16 cast
// ---------------------------------------------------------------------------
__global__ __launch_bounds__(256) void cast_kernel(const float* __restrict__ src,
                                                   u16* __restrict__ dst, int n) {
    int i = blockIdx.x * 256 + threadIdx.x;
    if (i < n) dst[i] = f2bf(src[i]);
}

// ---------------------------------------------------------------------------
// Tiled transpose + cast: src (R x C, f32, row-major) -> dst (C x R, bf16)
// ---------------------------------------------------------------------------
__global__ __launch_bounds__(256) void tcast_kernel(const float* __restrict__ src,
                                                    u16* __restrict__ dst,
                                                    int R, int C) {
    __shared__ float tile[32][33];
    const int tx = threadIdx.x & 31;
    const int ty = threadIdx.x >> 5;           // 0..7
    const int bx = blockIdx.x * 32;            // col base in src
    const int by = blockIdx.y * 32;            // row base in src
#pragma unroll
    for (int i = 0; i < 32; i += 8)
        tile[ty + i][tx] = src[(size_t)(by + ty + i) * C + bx + tx];
    __syncthreads();
#pragma unroll
    for (int i = 0; i < 32; i += 8)
        dst[(size_t)(bx + ty + i) * R + by + tx] = f2bf(tile[tx][ty + i]);
}

// ---------------------------------------------------------------------------
// LayerNorm over last dim (D=1024) -> bf16.  One block per row, 256 threads.
// ---------------------------------------------------------------------------
__global__ __launch_bounds__(256) void ln_kernel(const float* __restrict__ x,
                                                 const float* __restrict__ g,
                                                 const float* __restrict__ bb,
                                                 u16* __restrict__ out) {
    __shared__ float s1[256], s2[256];
    const int tid = threadIdx.x;
    const float* xr = x + (size_t)blockIdx.x * DM;
    float v[4], s = 0.f, ss = 0.f;
#pragma unroll
    for (int i = 0; i < 4; ++i) {
        v[i] = xr[tid + 256 * i];
        s += v[i]; ss += v[i] * v[i];
    }
    s1[tid] = s; s2[tid] = ss;
    __syncthreads();
    for (int o = 128; o > 0; o >>= 1) {
        if (tid < o) { s1[tid] += s1[tid + o]; s2[tid] += s2[tid + o]; }
        __syncthreads();
    }
    const float mu   = s1[0] * (1.0f / DM);
    const float var  = s2[0] * (1.0f / DM) - mu * mu;
    const float rstd = rsqrtf(var + 1e-5f);
    u16* orow = out + (size_t)blockIdx.x * DM;
#pragma unroll
    for (int i = 0; i < 4; ++i) {
        int c = tid + 256 * i;
        orow[c] = f2bf((v[i] - mu) * rstd * g[c] + bb[c]);
    }
}

// ---------------------------------------------------------------------------
// NT GEMM: C[m,n] = sum_k A[m,k] * W[n,k]   (A: MxK bf16, W: NxK bf16)
// Block tile 128x128 (8 waves as 4x2, each wave 32x64 via 2x4 WMMA frags),
// BK = 32.  Double-buffered LDS staged with GLOBAL_LOAD_ASYNC_TO_LDS_B128:
//   iter t: wait own ASYNCcnt, barrier (tile t resident everywhere, and all
//   readers of buf[t+1 mod 2] from iter t-1 are done), issue async copies of
//   tile t+1 into the other buffer, then WMMA on tile t.
// Epilogue modes:
//   0: bf16 plain store            (outB[m*N+n])
//   1: bf16 transposed "Vt" store  (outB[((m/Tp)*N + n)*Tp + m%Tp])
//   2: bf16 gelu(acc + bias[n])
//   3: f32  resid[m*N+n] + acc (+ bias[n] if bias != null) -> outF
// ---------------------------------------------------------------------------
__global__ __launch_bounds__(256) void gemm_nt_kernel(
        const u16* __restrict__ A, const u16* __restrict__ W,
        u16* __restrict__ outB, float* __restrict__ outF,
        const float* __restrict__ resid, const float* __restrict__ bias,
        int M, int N, int K, int mode, int Tp) {
    __shared__ __align__(16) u16 sA[2][128 * 40];   // 128 rows x 32 (+8 pad)
    __shared__ __align__(16) u16 sW[2][128 * 40];

    const int tid  = threadIdx.x;
    const int lane = tid & 31;
    const int wv   = tid >> 5;         // 0..7
    const int wr   = wv & 3;           // wave row (4)
    const int wc   = wv >> 2;          // wave col (2)
    const int bm   = blockIdx.y * 128;
    const int bn   = blockIdx.x * 128;
    const int ln16 = lane & 15;
    const int hi   = lane >> 4;        // 0 | 1

    v8f acc[2][4] = {};

    // staging: 256 threads cover 128 rows x 32 halves for each of A, W
    const int srow = tid >> 1, scol = (tid & 1) * 16;
    const u16* gA = A + (size_t)(bm + srow) * K + scol;
    const u16* gW = W + (size_t)(bn + srow) * K + scol;
    u16* lA = &sA[0][srow * 40 + scol];
    u16* lW = &sW[0][srow * 40 + scol];
    const size_t BUF = 128 * 40;       // halves per buffer

    // preload tile 0 into buffer 0
    async_g2l_b128(lA,     gA);
    async_g2l_b128(lA + 8, gA + 8);
    async_g2l_b128(lW,     gW);
    async_g2l_b128(lW + 8, gW + 8);

    const int nk = K / 32;
    for (int t = 0; t < nk; ++t) {
        wait_async0();                 // my async copies of tile t landed
        __syncthreads();               // everyone's landed; prev readers done
        if (t + 1 < nk) {              // stage tile t+1 into the other buffer
            const size_t ko = (size_t)(t + 1) * 32;
            const size_t bo = ((t + 1) & 1) * BUF;
            async_g2l_b128(lA + bo,     gA + ko);
            async_g2l_b128(lA + bo + 8, gA + ko + 8);
            async_g2l_b128(lW + bo,     gW + ko);
            async_g2l_b128(lW + bo + 8, gW + ko + 8);
        }
        const u16* cA = sA[t & 1];
        const u16* cW = sW[t & 1];

        Frag a0, a1, b[4];
        const int rA = wr * 32 + ln16;
        const int ka = hi * 8;         // A frag: lanes>=16 hold K 8..15 / 24..31
        a0.h[0] = *(const u16x8*)&cA[(rA)      * 40 + ka];
        a0.h[1] = *(const u16x8*)&cA[(rA)      * 40 + ka + 16];
        a1.h[0] = *(const u16x8*)&cA[(rA + 16) * 40 + ka];
        a1.h[1] = *(const u16x8*)&cA[(rA + 16) * 40 + ka + 16];
        const int nB = wc * 64 + ln16;
        const int kB = hi * 8;         // B frag: lanes>=16 hold K 8..15 / 24..31
#pragma unroll
        for (int j = 0; j < 4; ++j) {
            b[j].h[0] = *(const u16x8*)&cW[(nB + j * 16) * 40 + kB];
            b[j].h[1] = *(const u16x8*)&cW[(nB + j * 16) * 40 + kB + 16];
        }
#pragma unroll
        for (int j = 0; j < 4; ++j) {
            acc[0][j] = wmma_bf16(a0, b[j], acc[0][j]);
            acc[1][j] = wmma_bf16(a1, b[j], acc[1][j]);
        }
    }

    // epilogue: C-frag layout -> lane n = lane%16, row = e + 8*(lane>=16)
#pragma unroll
    for (int i = 0; i < 2; ++i)
#pragma unroll
        for (int j = 0; j < 4; ++j) {
            const int mbase = bm + wr * 32 + i * 16 + hi * 8;
            const int nn    = bn + wc * 64 + j * 16 + ln16;
#pragma unroll
            for (int e = 0; e < 8; ++e) {
                const int m = mbase + e;
                float v = acc[i][j][e];
                if (mode == 0) {
                    outB[(size_t)m * N + nn] = f2bf(v);
                } else if (mode == 1) {
                    outB[((size_t)(m / Tp) * N + nn) * Tp + (m % Tp)] = f2bf(v);
                } else if (mode == 2) {
                    outB[(size_t)m * N + nn] = f2bf(gelu_f(v + bias[nn]));
                } else {
                    if (bias) v += bias[nn];
                    outF[(size_t)m * N + nn] = resid[(size_t)m * N + nn] + v;
                }
            }
        }
}

// ---------------------------------------------------------------------------
// Flash attention: one wave per 16-query tile; K-tiles of 32; online softmax.
// Q,K: (B*T, D) bf16 per-head slices (row stride D); Vt: (B, D, T) bf16.
// ---------------------------------------------------------------------------
__global__ __launch_bounds__(128) void attn_kernel(const u16* __restrict__ Q,
                                                   const u16* __restrict__ Kc,
                                                   const u16* __restrict__ Vt,
                                                   u16* __restrict__ Oa) {
    __shared__ __align__(16) u16 sp[4][16 * 32];   // per-wave P staging
    const int lane = threadIdx.x & 31;
    const int wv   = threadIdx.x >> 5;
    const int ln16 = lane & 15;
    const int hi   = lane >> 4;

    const int qt   = blockIdx.x * 4 + wv;          // 0..4095
    const int tph  = TT / 16;                      // tiles per head = 128
    const int hl   = qt / tph;                     // 0..31 (b*NH + h)
    const int q0   = (qt % tph) * 16;
    const int b    = hl / NH;
    const int h    = hl % NH;

    const u16* qp = Q  + (size_t)b * TT * DM + h * HD;
    const u16* kp = Kc + (size_t)b * TT * DM + h * HD;
    const u16* vp = Vt + ((size_t)b * DM + h * HD) * TT;
    u16*       op = Oa + (size_t)b * TT * DM + h * HD;

    // Q A-fragments (row q0+ln16, two K=32 chunks of the 64 head dims)
    Frag aq[2];
    {
        const u16* qr = qp + (size_t)(q0 + ln16) * DM;
#pragma unroll
        for (int dk = 0; dk < 2; ++dk) {
            const int kb = dk * 32 + hi * 8;
            aq[dk].h[0] = *(const u16x8*)(qr + kb);
            aq[dk].h[1] = *(const u16x8*)(qr + kb + 16);
        }
    }

    float mrow[8], lrow[8];
    v8f o0 = {}, o1 = {}, o2 = {}, o3 = {};
#pragma unroll
    for (int j = 0; j < 8; ++j) { mrow[j] = -1e30f; lrow[j] = 0.f; }

    const int nkt = (q0 + 15) / 32 + 1;
    for (int kt = 0; kt < nkt; ++kt) {
        const int kk0 = kt * 32;
        v8f s0 = {}, s1 = {};
#pragma unroll
        for (int dk = 0; dk < 2; ++dk) {           // d chunks 0..31 / 32..63
            const int db = dk * 32 + hi * 8;
            Frag bk0, bk1;
            const u16* k0r = kp + (size_t)(kk0 + ln16)      * DM + db;
            const u16* k1r = kp + (size_t)(kk0 + 16 + ln16) * DM + db;
            bk0.h[0] = *(const u16x8*)k0r;  bk0.h[1] = *(const u16x8*)(k0r + 16);
            bk1.h[0] = *(const u16x8*)k1r;  bk1.h[1] = *(const u16x8*)(k1r + 16);
            s0 = wmma_bf16(aq[dk], bk0, s0);
            s1 = wmma_bf16(aq[dk], bk1, s1);
        }

        // scale + causal mask + online softmax (per-row stats across 16 lanes)
        float p0[8], p1[8], alpha[8];
#pragma unroll
        for (int j = 0; j < 8; ++j) {
            const int row = q0 + j + hi * 8;
            const int c0  = kk0 + ln16;
            const int c1  = kk0 + 16 + ln16;
            float v0 = s0[j] * 0.125f;
            float v1 = s1[j] * 0.125f;
            if (c0 > row) v0 = -1e30f;
            if (c1 > row) v1 = -1e30f;
            float tm = fmaxf(v0, v1);
#pragma unroll
            for (int off = 8; off > 0; off >>= 1)
                tm = fmaxf(tm, __shfl_xor(tm, off, 32));
            const float mnew = fmaxf(mrow[j], tm);
            alpha[j] = __expf(mrow[j] - mnew);
            p0[j] = __expf(v0 - mnew);
            p1[j] = __expf(v1 - mnew);
            float rs = p0[j] + p1[j];
#pragma unroll
            for (int off = 8; off > 0; off >>= 1)
                rs += __shfl_xor(rs, off, 32);
            lrow[j] = lrow[j] * alpha[j] + rs;
            mrow[j] = mnew;
        }
#pragma unroll
        for (int j = 0; j < 8; ++j) {
            o0[j] *= alpha[j]; o1[j] *= alpha[j];
            o2[j] *= alpha[j]; o3[j] *= alpha[j];
        }

        // P (C layout) -> LDS -> A-fragment (same-wave LDS is in-order)
        u16* pl = sp[wv];
#pragma unroll
        for (int j = 0; j < 8; ++j) {
            const int m = j + hi * 8;
            pl[m * 32 + ln16]      = f2bf(p0[j]);
            pl[m * 32 + 16 + ln16] = f2bf(p1[j]);
        }
        asm volatile("s_wait_dscnt 0" ::: "memory");
        Frag pf;
        {
            const int kb = hi * 8;
            pf.h[0] = *(u16x8*)&pl[ln16 * 32 + kb];
            pf.h[1] = *(u16x8*)&pl[ln16 * 32 + kb + 16];
        }

        // O += P @ V   (Vt row-contiguous over t)
#pragma unroll
        for (int dc = 0; dc < 4; ++dc) {
            Frag bv;
            const u16* vr = vp + (size_t)(dc * 16 + ln16) * TT + kk0 + hi * 8;
            bv.h[0] = *(const u16x8*)vr;
            bv.h[1] = *(const u16x8*)(vr + 16);
            v8f& od = (dc == 0) ? o0 : (dc == 1) ? o1 : (dc == 2) ? o2 : o3;
            od = wmma_bf16(pf, bv, od);
        }
    }

    // normalize + store (B*T, D) bf16
#pragma unroll
    for (int j = 0; j < 8; ++j) {
        const float inv = 1.0f / lrow[j];
        u16* orow = op + (size_t)(q0 + j + hi * 8) * DM;
        orow[0 * 16 + ln16] = f2bf(o0[j] * inv);
        orow[1 * 16 + ln16] = f2bf(o1[j] * inv);
        orow[2 * 16 + ln16] = f2bf(o2[j] * inv);
        orow[3 * 16 + ln16] = f2bf(o3[j] * inv);
    }
}

// ---------------------------------------------------------------------------
// Host-side orchestration
// ---------------------------------------------------------------------------
extern "C" void kernel_launch(void* const* d_in, const int* in_sizes, int n_in,
                              void* d_out, int out_size, void* d_ws, size_t ws_size,
                              hipStream_t stream) {
    const float* x    = (const float*)d_in[0];
    const float* wq   = (const float*)d_in[1];
    const float* wk   = (const float*)d_in[2];
    const float* wvv  = (const float*)d_in[3];
    const float* wo   = (const float*)d_in[4];
    const float* ln1g = (const float*)d_in[5];
    const float* ln1b = (const float*)d_in[6];
    const float* ln2g = (const float*)d_in[7];
    const float* ln2b = (const float*)d_in[8];
    const float* w1   = (const float*)d_in[9];
    const float* b1   = (const float*)d_in[10];
    const float* w2   = (const float*)d_in[11];
    const float* b2   = (const float*)d_in[12];
    float* out        = (float*)d_out;

    char* ws = (char*)d_ws;
    const size_t MB = 1024u * 1024u;
    u16*   h_bf   = (u16*)(ws + 0 * MB);     // (M, D)        8 MB
    u16*   wq_bf  = (u16*)(ws + 8 * MB);     // (D, D)        2 MB
    u16*   wk_bf  = (u16*)(ws + 10 * MB);
    u16*   wv_bf  = (u16*)(ws + 12 * MB);
    u16*   wo_bf  = (u16*)(ws + 14 * MB);
    u16*   w1t_bf = (u16*)(ws + 16 * MB);    // (DFF, D)      8 MB
    u16*   w2t_bf = (u16*)(ws + 24 * MB);    // (D, DFF)      8 MB
    u16*   q_bf   = (u16*)(ws + 32 * MB);    // (M, D)        8 MB
    u16*   k_bf   = (u16*)(ws + 40 * MB);    // (M, D)        8 MB
    u16*   vt_bf  = (u16*)(ws + 48 * MB);    // (B, D, T)     8 MB
    u16*   att_bf = (u16*)(ws + 56 * MB);    // (M, D)        8 MB
    float* x1     = (float*)(ws + 64 * MB);  // (M, D) f32   16 MB
    u16*   h2_bf  = (u16*)(ws + 80 * MB);    // (M, D)        8 MB
    u16*   ff_bf  = (u16*)(ws + 88 * MB);    // (M, DFF)     32 MB

    // 1) weight casts / transposes
    cast_kernel<<<(DM * DM) / 256, 256, 0, stream>>>(wq,  wq_bf, DM * DM);
    cast_kernel<<<(DM * DM) / 256, 256, 0, stream>>>(wk,  wk_bf, DM * DM);
    cast_kernel<<<(DM * DM) / 256, 256, 0, stream>>>(wvv, wv_bf, DM * DM);
    cast_kernel<<<(DM * DM) / 256, 256, 0, stream>>>(wo,  wo_bf, DM * DM);
    tcast_kernel<<<dim3(DFF / 32, DM / 32), 256, 0, stream>>>(w1, w1t_bf, DM, DFF);
    tcast_kernel<<<dim3(DM / 32, DFF / 32), 256, 0, stream>>>(w2, w2t_bf, DFF, DM);

    // 2) LN1
    ln_kernel<<<MROWS, 256, 0, stream>>>(x, ln1g, ln1b, h_bf);

    // 3) QKV projections (NT GEMMs); V stored transposed per (b, d) row
    gemm_nt_kernel<<<dim3(DM / 128, MROWS / 128), 256, 0, stream>>>(
        h_bf, wq_bf, q_bf, nullptr, nullptr, nullptr, MROWS, DM, DM, 0, TT);
    gemm_nt_kernel<<<dim3(DM / 128, MROWS / 128), 256, 0, stream>>>(
        h_bf, wk_bf, k_bf, nullptr, nullptr, nullptr, MROWS, DM, DM, 0, TT);
    gemm_nt_kernel<<<dim3(DM / 128, MROWS / 128), 256, 0, stream>>>(
        h_bf, wv_bf, vt_bf, nullptr, nullptr, nullptr, MROWS, DM, DM, 1, TT);

    // 4) flash attention: 4096 q-tiles, 4 waves/block
    attn_kernel<<<(TB * NH * (TT / 16)) / 4, 128, 0, stream>>>(q_bf, k_bf, vt_bf, att_bf);

    // 5) x1 = x + attn @ wo.T (f32)
    gemm_nt_kernel<<<dim3(DM / 128, MROWS / 128), 256, 0, stream>>>(
        att_bf, wo_bf, nullptr, x1, x, nullptr, MROWS, DM, DM, 3, TT);

    // 6) LN2
    ln_kernel<<<MROWS, 256, 0, stream>>>(x1, ln2g, ln2b, h2_bf);

    // 7) ff = gelu(h2 @ w1 + b1)
    gemm_nt_kernel<<<dim3(DFF / 128, MROWS / 128), 256, 0, stream>>>(
        h2_bf, w1t_bf, ff_bf, nullptr, nullptr, b1, MROWS, DFF, DM, 2, TT);

    // 8) out = x1 + ff @ w2 + b2 (f32)
    gemm_nt_kernel<<<dim3(DM / 128, MROWS / 128), 256, 0, stream>>>(
        ff_bf, w2t_bf, nullptr, out, x1, b2, MROWS, DM, DFF, 3, TT);

    (void)in_sizes; (void)n_in; (void)out_size; (void)ws_size;
}
